// RIMCell_19507741458958
// MI455X (gfx1250) — matrix operand order repaired and checked by reading, not compiled, not run on hardware
//
#include <hip/hip_runtime.h>
#include <hip/hip_bf16.h>
#include <math.h>

// ---------------- problem constants ----------------
constexpr int Dd   = 600;
constexpr int NUc  = 8;
constexpr int Hc   = 512;
constexpr int KTOPc= 5;
constexpr int IKDc = 64;
constexpr int IVDc = 400;
constexpr int IVDP = 416;          // IVD padded to a multiple of 32 for the WMMA K loop
constexpr int NCHc = 4;
constexpr int CKDc = 32;
constexpr int CVDc = 512;
constexpr int G3H  = 3 * Hc;       // 1536

// ---------------- WMMA types ----------------
typedef __attribute__((ext_vector_type(16))) __bf16        v16bf;
typedef __attribute__((ext_vector_type(8)))  float         v8f;
typedef __attribute__((ext_vector_type(8)))  unsigned int  v8u;

__device__ __forceinline__ unsigned short f2bf(float f) {
  unsigned int u = __builtin_bit_cast(unsigned int, f);
  u += 0x7FFFu + ((u >> 16) & 1u);               // round-to-nearest-even
  return (unsigned short)(u >> 16);
}
__device__ __forceinline__ float bf2f(unsigned short s) {
  return __builtin_bit_cast(float, (unsigned int)s << 16);
}

// ---------------- fragment-swizzled operand layouts (ISA 7.12.2) ----------------
// Tiles of 512 ushorts = 32 lanes x 8 dwords, each lane's 8 dwords contiguous (32B).
// A (M x K): tile = (m>>4)*(K>>5) + (k>>5)   (K-tiles contiguous per row-tile)
//   lane = (m&15) | hi<<4  with hi=(k>>3)&1 ; vgpr v = ((k&31)>>4)*4 + ((k&7)>>1)
__device__ __forceinline__ long long a_sw_idx(long long m, int k, int K) {
  const long long tile = (m >> 4) * (long long)(K >> 5) + (k >> 5);
  const int kk = k & 31;
  const int hi = (kk >> 3) & 1;
  const int v  = ((kk >> 4) << 2) + ((kk & 7) >> 1);
  const int lane = (int)(m & 15) | (hi << 4);
  return tile * 512 + lane * 16 + v * 2 + (kk & 1);
}
// B (K x N): tile = (n>>4)*(K>>5) + (k>>5)   (K-tiles contiguous per column-tile)
//   lane = (n&15) | ((k>>3)&1)<<4 ; vgpr v = ((k&31)>>4)*4 + ((k&7)>>1)
__device__ __forceinline__ long long b_sw_idx(int k, int n, int K, int N) {
  const long long tile = (long long)(n >> 4) * (K >> 5) + (k >> 5);
  const int kk = k & 31;
  const int lane = (n & 15) | (((kk >> 3) & 1) << 4);
  const int v = ((kk >> 4) << 2) + ((kk & 7) >> 1);
  return tile * 512 + lane * 16 + v * 2 + (kk & 1);
}

__device__ __forceinline__ v16bf load_frag_sw(const unsigned short* p) {
  const v8u t = *(const v8u*)p;                  // 32B contiguous -> 2x global_load_b128
  return __builtin_bit_cast(v16bf, t);
}

__device__ __forceinline__ v8f wmma_bf16(v16bf a, v16bf b, v8f c) {
  return __builtin_amdgcn_wmma_f32_16x16x32_bf16(false, a, false, b, (short)0, c,
                                                 false, false);
}

// ---------------- converters into swizzled layouts ----------------
// U matrices (Ksrc x N) f32 row-major -> swizzled bf16 (Kdst x N), zero-padded K rows
__global__ void cvt_w_sw_kernel(const float* __restrict__ src,
                                unsigned short* __restrict__ dst,
                                int U, int Ksrc, int Kdst, int N) {
  const long long i = (long long)blockIdx.x * blockDim.x + threadIdx.x;
  const long long tot = (long long)U * Kdst * N;
  if (i >= tot) return;
  const int n = (int)(i % N);
  const long long t = i / N;
  const int k = (int)(t % Kdst);
  const int u = (int)(t / Kdst);
  const float v = (k < Ksrc) ? src[((long long)u * Ksrc + k) * N + n] : 0.f;
  dst[(long long)u * ((long long)Kdst * N) + b_sw_idx(k, n, Kdst, N)] = f2bf(v);
}

// hs (B,NU,H) f32 -> per-unit swizzled A matrices (B x H)
__global__ void cvt_hs_sw_kernel(const float* __restrict__ src,
                                 unsigned short* __restrict__ dst, long long Bsz) {
  const long long i = (long long)blockIdx.x * blockDim.x + threadIdx.x;
  const long long tot = Bsz * NUc * Hc;
  if (i >= tot) return;
  const int k = (int)(i % Hc);
  const long long t = i / Hc;
  const int u = (int)(t % NUc);
  const long long b = t / NUc;
  dst[(long long)u * (Bsz * Hc) + a_sw_idx(b, k, Hc)] = f2bf(src[i]);
}

// ---------------- stage 1: input attention + top-k mask ----------------
__global__ void __launch_bounds__(256)
stage1_kernel(const float* __restrict__ x, const float* __restrict__ hs,
              const float* __restrict__ Wk, const float* __restrict__ bk,
              const float* __restrict__ Wv, const float* __restrict__ bv,
              const float* __restrict__ Wq, const float* __restrict__ bq,
              float* __restrict__ maskOut, unsigned short* __restrict__ inSw,
              long long Bsz) {
  __shared__ float xr[Dd];
  __shared__ float hsr[NUc * Hc];
  __shared__ float kk0[IKDc];
  __shared__ float vv0[IVDc];
  __shared__ float qq[NUc * IKDc];
  __shared__ float sc[NUc][2];
  __shared__ float pb[NUc][2];
  __shared__ float msk[NUc];
  const int b = blockIdx.x, tid = threadIdx.x;

  for (int i = tid; i < Dd; i += 256) xr[i] = x[(long long)b * Dd + i];
  for (int i = tid; i < NUc * Hc; i += 256) hsr[i] = hs[(long long)b * NUc * Hc + i];
  __syncthreads();

  for (int j = tid; j < IKDc; j += 256) {       // kk row 0 (xc row 1 is zeros -> bias)
    float s = bk[j];
    for (int d = 0; d < Dd; ++d) s += xr[d] * Wk[(long long)d * IKDc + j];
    kk0[j] = s;
  }
  for (int c = tid; c < IVDc; c += 256) {       // vv row 0
    float s = bv[c];
    for (int d = 0; d < Dd; ++d) s += xr[d] * Wv[(long long)d * IVDc + c];
    vv0[c] = s;
  }
  for (int t = tid; t < NUc * IKDc; t += 256) { // qq
    const int u = t / IKDc, j = t % IKDc;
    float s = bq[j];
    for (int h = 0; h < Hc; ++h) s += hsr[u * Hc + h] * Wq[(long long)h * IKDc + j];
    qq[t] = s;
  }
  __syncthreads();

  if (tid < NUc) {                              // scores + softmax over 2 slots
    const int u = tid;
    float s0 = 0.f, s1 = 0.f;
    for (int j = 0; j < IKDc; ++j) {
      s0 += qq[u * IKDc + j] * kk0[j];
      s1 += qq[u * IKDc + j] * bk[j];           // kk row 1 = bias
    }
    s0 *= 0.125f; s1 *= 0.125f;                 // 1/sqrt(IKD)
    sc[u][0] = s0; sc[u][1] = s1;
    const float mx = fmaxf(s0, s1);
    const float e0 = __expf(s0 - mx), e1 = __expf(s1 - mx);
    const float inv = 1.f / (e0 + e1);
    pb[u][0] = e0 * inv; pb[u][1] = e1 * inv;
    msk[u] = 0.f;
  }
  __syncthreads();
  if (tid == 0) {                               // top-KTOP of scores[:,0], ties -> lower u
    bool used[NUc] = {};
    for (int t = 0; t < KTOPc; ++t) {
      float best = -INFINITY; int bi = 0;
      for (int u = 0; u < NUc; ++u)
        if (!used[u] && sc[u][0] > best) { best = sc[u][0]; bi = u; }
      used[bi] = true; msk[bi] = 1.f;
    }
  }
  __syncthreads();
  if (tid < NUc) maskOut[(long long)b * NUc + tid] = msk[tid];
  for (int t = tid; t < NUc * IVDP; t += 256) { // masked inputs, fragment-swizzled
    const int u = t / IVDP, c = t % IVDP;
    float val = 0.f;
    if (c < IVDc) val = (pb[u][0] * vv0[c] + pb[u][1] * bv[c]) * msk[u];
    inSw[(long long)u * (Bsz * IVDP) + a_sw_idx((long long)b, c, IVDP)] = f2bf(val);
  }
}

// ---------------- GRU: 6 fused WMMA GEMM tiles + gating epilogue ----------------
__global__ void __launch_bounds__(128)
gru_wmma_kernel(const unsigned short* __restrict__ inSw,   // per-u (B x 416) swizzled
                const unsigned short* __restrict__ hsSw,   // per-u (B x 512) swizzled
                const unsigned short* __restrict__ WihSw,  // per-u (416 x 1536) swizzled
                const unsigned short* __restrict__ WhhSw,  // per-u (512 x 1536) swizzled
                const float* __restrict__ b_ih, const float* __restrict__ b_hh,
                const float* __restrict__ hs,
                float* __restrict__ hbgF, unsigned short* __restrict__ hbgSw,
                long long Bsz) {
  const int lane = threadIdx.x & 31;
  const int wave = threadIdx.x >> 5;
  const long long b0 = (long long)blockIdx.x * 16;
  const int u  = blockIdx.y;
  const int nb = blockIdx.z * 64 + wave * 16;   // column sub-tile within H

  v8f a_ir = {}, a_iz = {}, a_in = {};
  {
    const int kt = IVDP >> 5;                   // 13
    const unsigned short* aP = inSw + (long long)u * (Bsz * IVDP)
                             + (long long)blockIdx.x * kt * 512 + lane * 16;
    const unsigned short* wB = WihSw + (long long)u * IVDP * G3H + lane * 16;
    const unsigned short* bPr = wB + (long long)(nb >> 4)            * kt * 512;
    const unsigned short* bPz = wB + (long long)((nb + Hc) >> 4)     * kt * 512;
    const unsigned short* bPn = wB + (long long)((nb + 2 * Hc) >> 4) * kt * 512;
    for (int t = 0; t < kt; ++t) {
      __builtin_prefetch(aP + 2048, 0, 3);
      const v16bf a = load_frag_sw(aP);
      a_ir = wmma_bf16(a, load_frag_sw(bPr), a_ir);
      a_iz = wmma_bf16(a, load_frag_sw(bPz), a_iz);
      a_in = wmma_bf16(a, load_frag_sw(bPn), a_in);
      aP += 512; bPr += 512; bPz += 512; bPn += 512;
    }
  }
  v8f a_hr = {}, a_hz = {}, a_hn = {};
  {
    const int kt = Hc >> 5;                     // 16
    const unsigned short* aP = hsSw + (long long)u * (Bsz * Hc)
                             + (long long)blockIdx.x * kt * 512 + lane * 16;
    const unsigned short* wB = WhhSw + (long long)u * Hc * G3H + lane * 16;
    const unsigned short* bPr = wB + (long long)(nb >> 4)            * kt * 512;
    const unsigned short* bPz = wB + (long long)((nb + Hc) >> 4)     * kt * 512;
    const unsigned short* bPn = wB + (long long)((nb + 2 * Hc) >> 4) * kt * 512;
    for (int t = 0; t < kt; ++t) {
      __builtin_prefetch(aP + 2048, 0, 3);
      const v16bf a = load_frag_sw(aP);
      a_hr = wmma_bf16(a, load_frag_sw(bPr), a_hr);
      a_hz = wmma_bf16(a, load_frag_sw(bPz), a_hz);
      a_hn = wmma_bf16(a, load_frag_sw(bPn), a_hn);
      aP += 512; bPr += 512; bPz += 512; bPn += 512;
    }
  }
  // Epilogue: C/D layout -> column j fixed per lane, row varies over the 8 VGPRs.
  const int j = nb + (lane & 15);
  const float bir = b_ih[u * G3H + j], biz = b_ih[u * G3H + Hc + j], bin = b_ih[u * G3H + 2 * Hc + j];
  const float bhr = b_hh[u * G3H + j], bhz = b_hh[u * G3H + Hc + j], bhn = b_hh[u * G3H + 2 * Hc + j];
#pragma unroll
  for (int r = 0; r < 8; ++r) {
    const long long bb  = b0 + r + (lane >> 4) * 8;
    const long long row = bb * NUc + u;
    const long long idx = row * Hc + j;
    const float hsv = hs[idx];
    const float rg  = 1.f / (1.f + __expf(-(a_ir[r] + bir + a_hr[r] + bhr)));
    const float zg  = 1.f / (1.f + __expf(-(a_iz[r] + biz + a_hz[r] + bhz)));
    const float ng  = tanhf(a_in[r] + bin + rg * (a_hn[r] + bhn));
    const float h   = (1.f - zg) * ng + zg * hsv; // h_rnn == h_bg in forward
    hbgF[idx] = h;
    hbgSw[a_sw_idx(row, j, Hc)] = f2bf(h);        // A operand for qc/kc/vc GEMMs
  }
}

// ---------------- generic swizzled bf16 WMMA GEMM: out = A(MxK) @ W(KxN) + bias ----
__global__ void __launch_bounds__(128)
gemm_bf16_wmma_sw(const unsigned short* __restrict__ A,   // swizzled (M x K)
                  const unsigned short* __restrict__ W,   // swizzled (K x N)
                  int N, int K, const float* __restrict__ bias,
                  float* __restrict__ outF, unsigned short* __restrict__ outB) {
  const int lane = threadIdx.x & 31;
  const int wave = threadIdx.x >> 5;
  const int kt = K >> 5;
  const long long m0 = (long long)blockIdx.x * 16;
  const int n0 = blockIdx.y * 64 + wave * 16;
  const unsigned short* aP = A + (long long)blockIdx.x * kt * 512 + lane * 16;
  const unsigned short* bP = W + (long long)(n0 >> 4) * kt * 512 + lane * 16;
  v8f acc = {};
  for (int t = 0; t < kt; ++t) {
    __builtin_prefetch(aP + 2048, 0, 3);
    __builtin_prefetch(bP + 2048, 0, 3);
    const v16bf a = load_frag_sw(aP);
    const v16bf b = load_frag_sw(bP);
    aP += 512; bP += 512;
    acc = wmma_bf16(a, b, acc);
  }
  const int col = n0 + (lane & 15);
  const float bv = bias ? bias[col] : 0.f;
#pragma unroll
  for (int r = 0; r < 8; ++r) {
    const long long row = m0 + r + (lane >> 4) * 8;
    const float val = acc[r] + bv;
    if (outF) outF[row * (long long)N + col] = val;      // row-major f32
    if (outB) outB[row * (long long)N + col] = f2bf(val); // row-major bf16 (vc)
  }
}

// ---------------- per-head 8x8 context attention ----------------
__global__ void __launch_bounds__(256)
ctx_attn_kernel(const float* __restrict__ qc, const float* __restrict__ kc,
                const unsigned short* __restrict__ vcBf, const float* __restrict__ mask,
                unsigned short* __restrict__ ctxSw) {
  __shared__ float q_s[NUc][CKDc], k_s[NUc][CKDc], s_s[NUc][NUc], p_s[NUc][NUc];
  const int b = blockIdx.x, h = blockIdx.y, tid = threadIdx.x;
  for (int t = tid; t < NUc * CKDc; t += 256) {
    const int u = t / CKDc, k = t % CKDc;
    q_s[u][k] = qc[((long long)b * NUc + u) * (NCHc * CKDc) + h * CKDc + k];
    k_s[u][k] = kc[((long long)b * NUc + u) * (NCHc * CKDc) + h * CKDc + k];
  }
  __syncthreads();
  if (tid < NUc * NUc) {
    const int u = tid / NUc, v = tid % NUc;
    float s = 0.f;
    for (int k = 0; k < CKDc; ++k) s += q_s[u][k] * k_s[v][k];
    s_s[u][v] = s * 0.17677669529663687f;       // 1/sqrt(32)
  }
  __syncthreads();
  if (tid < NUc) {
    const int u = tid;
    float mx = -INFINITY;
    for (int v = 0; v < NUc; ++v) mx = fmaxf(mx, s_s[u][v]);
    float e[NUc], sum = 0.f;
    for (int v = 0; v < NUc; ++v) { e[v] = __expf(s_s[u][v] - mx); sum += e[v]; }
    const float m = mask[(long long)b * NUc + u] / sum;
    for (int v = 0; v < NUc; ++v) p_s[u][v] = e[v] * m;
  }
  __syncthreads();
  for (int t = tid; t < NUc * CVDc; t += 256) { // ctx, written fragment-swizzled
    const int u = t / CVDc, c = t % CVDc;
    float s = 0.f;
    for (int v = 0; v < NUc; ++v)
      s += p_s[u][v] * bf2f(vcBf[((long long)b * NUc + v) * (NCHc * CVDc) + h * CVDc + c]);
    ctxSw[a_sw_idx((long long)b * NUc + u, h * CVDc + c, NCHc * CVDc)] = f2bf(s);
  }
}

// ---------------- final blend: out = m*(h_bg + ctx) + (1-m)*hs ----------------
__global__ void blend_kernel(const float* __restrict__ hbg, const float* __restrict__ ctxo,
                             const float* __restrict__ hs, const float* __restrict__ mask,
                             float* __restrict__ out, long long n) {
  const long long i = (long long)blockIdx.x * blockDim.x + threadIdx.x;
  if (i >= n) return;
  const long long row = i / Hc;                 // b*NU + u
  const float m = mask[row];
  out[i] = m * (hbg[i] + ctxo[i]) + (1.f - m) * hs[i];
}

// ---------------- host launch ----------------
extern "C" void kernel_launch(void* const* d_in, const int* in_sizes, int n_in,
                              void* d_out, int out_size, void* d_ws, size_t ws_size,
                              hipStream_t stream) {
  const float* x     = (const float*)d_in[0];
  const float* hs    = (const float*)d_in[1];
  const float* Wk_in = (const float*)d_in[2];
  const float* bk_in = (const float*)d_in[3];
  const float* Wv_in = (const float*)d_in[4];
  const float* bv_in = (const float*)d_in[5];
  const float* Wq_in = (const float*)d_in[6];
  const float* bq_in = (const float*)d_in[7];
  const float* W_ih  = (const float*)d_in[8];
  const float* W_hh  = (const float*)d_in[9];
  const float* b_ih  = (const float*)d_in[10];
  const float* b_hh  = (const float*)d_in[11];
  const float* Wq_c  = (const float*)d_in[12];
  const float* bq_c  = (const float*)d_in[13];
  const float* Wk_c  = (const float*)d_in[14];
  const float* bk_c  = (const float*)d_in[15];
  const float* Wv_c  = (const float*)d_in[16];
  const float* bv_c  = (const float*)d_in[17];
  const float* W_out = (const float*)d_in[18];
  const float* b_out = (const float*)d_in[19];
  float* out = (float*)d_out;

  const long long Bsz = in_sizes[0] / Dd;       // 2048
  const long long M   = Bsz * NUc;              // 16384 flattened rows

  // workspace carve-up (256B aligned)
  char* p = (char*)d_ws;
  auto alloc = [&](size_t bytes) -> char* {
    char* r = p; p += (bytes + 255) & ~(size_t)255; return r;
  };
  float*          maskW  = (float*)         alloc(sizeof(float) * Bsz * NUc);
  unsigned short* inSw   = (unsigned short*)alloc(2ull * Bsz * NUc * IVDP);
  unsigned short* hsSw   = (unsigned short*)alloc(2ull * Bsz * NUc * Hc);
  unsigned short* WihSw  = (unsigned short*)alloc(2ull * NUc * IVDP * G3H);
  unsigned short* WhhSw  = (unsigned short*)alloc(2ull * NUc * Hc * G3H);
  unsigned short* WqcSw  = (unsigned short*)alloc(2ull * Hc * NCHc * CKDc);
  unsigned short* WkcSw  = (unsigned short*)alloc(2ull * Hc * NCHc * CKDc);
  unsigned short* WvcSw  = (unsigned short*)alloc(2ull * Hc * NCHc * CVDc);
  unsigned short* WoutSw = (unsigned short*)alloc(2ull * NCHc * CVDc * CVDc);
  float*          hbgF   = (float*)         alloc(4ull * Bsz * NUc * Hc);
  unsigned short* hbgSw  = (unsigned short*)alloc(2ull * Bsz * NUc * Hc);
  float*          qcF    = (float*)         alloc(4ull * M * NCHc * CKDc);
  float*          kcF    = (float*)         alloc(4ull * M * NCHc * CKDc);
  unsigned short* vcBf   = (unsigned short*)alloc(2ull * M * NCHc * CVDc);
  unsigned short* ctxSw  = (unsigned short*)alloc(2ull * M * NCHc * CVDc);
  float*          ctxoF  = (float*)         alloc(4ull * M * CVDc);

  auto nblk = [](long long n) { return (unsigned)((n + 255) / 256); };

  // 1) weight + hs conversion into fragment-swizzled bf16
  cvt_w_sw_kernel<<<nblk((long long)NUc * IVDP * G3H), 256, 0, stream>>>(
      W_ih, WihSw, NUc, IVDc, IVDP, G3H);                      // zero-pad K 400->416
  cvt_w_sw_kernel<<<nblk((long long)NUc * Hc * G3H), 256, 0, stream>>>(
      W_hh, WhhSw, NUc, Hc, Hc, G3H);
  cvt_w_sw_kernel<<<nblk((long long)Hc * NCHc * CKDc), 256, 0, stream>>>(
      Wq_c, WqcSw, 1, Hc, Hc, NCHc * CKDc);
  cvt_w_sw_kernel<<<nblk((long long)Hc * NCHc * CKDc), 256, 0, stream>>>(
      Wk_c, WkcSw, 1, Hc, Hc, NCHc * CKDc);
  cvt_w_sw_kernel<<<nblk((long long)Hc * NCHc * CVDc), 256, 0, stream>>>(
      Wv_c, WvcSw, 1, Hc, Hc, NCHc * CVDc);
  cvt_w_sw_kernel<<<nblk((long long)NCHc * CVDc * CVDc), 256, 0, stream>>>(
      W_out, WoutSw, 1, NCHc * CVDc, NCHc * CVDc, CVDc);
  cvt_hs_sw_kernel<<<nblk(Bsz * NUc * Hc), 256, 0, stream>>>(hs, hsSw, Bsz);

  // 2) input attention + top-k mask -> masked inputs (swizzled bf16)
  stage1_kernel<<<(unsigned)Bsz, 256, 0, stream>>>(x, hs, Wk_in, bk_in, Wv_in, bv_in,
                                                   Wq_in, bq_in, maskW, inSw, Bsz);

  // 3) GRU: 6 WMMA GEMM tiles fused with gating  (grid: B/16 x NU x H/64)
  gru_wmma_kernel<<<dim3((unsigned)(Bsz / 16), NUc, Hc / 64), 128, 0, stream>>>(
      inSw, hsSw, WihSw, WhhSw, b_ih, b_hh, hs, hbgF, hbgSw, Bsz);

  // 4) qc / kc / vc projections (rows = B*NU)
  gemm_bf16_wmma_sw<<<dim3((unsigned)(M / 16), (NCHc * CKDc) / 64), 128, 0, stream>>>(
      hbgSw, WqcSw, NCHc * CKDc, Hc, bq_c, qcF, nullptr);
  gemm_bf16_wmma_sw<<<dim3((unsigned)(M / 16), (NCHc * CKDc) / 64), 128, 0, stream>>>(
      hbgSw, WkcSw, NCHc * CKDc, Hc, bk_c, kcF, nullptr);
  gemm_bf16_wmma_sw<<<dim3((unsigned)(M / 16), (NCHc * CVDc) / 64), 128, 0, stream>>>(
      hbgSw, WvcSw, NCHc * CVDc, Hc, bv_c, nullptr, vcBf);

  // 5) per-head softmax attention -> ctx (swizzled bf16)
  ctx_attn_kernel<<<dim3((unsigned)Bsz, NCHc), 256, 0, stream>>>(qcF, kcF, vcBf, maskW, ctxSw);

  // 6) output projection: ctx (M x 2048) @ W_out (2048 x 512)
  gemm_bf16_wmma_sw<<<dim3((unsigned)(M / 16), CVDc / 64), 128, 0, stream>>>(
      ctxSw, WoutSw, CVDc, NCHc * CVDc, b_out, ctxoF, nullptr);

  // 7) final blend
  blend_kernel<<<nblk(Bsz * NUc * Hc), 256, 0, stream>>>(hbgF, ctxoF, hs, maskW, out,
                                                         Bsz * NUc * Hc);
  (void)n_in; (void)out_size; (void)ws_size;
}
